// CrossAttention_28681791602801
// MI455X (gfx1250) — compile-verified, matrix operand order checked by reading, not run on hardware
//
#include <hip/hip_runtime.h>
#include <hip/hip_bf16.h>
#include <cmath>

// ---------------------------------------------------------------------------
// CDNA5 (gfx1250) self-attention, f16 WMMA path, cross-iteration pipelined.
//   B=2, S=4096, QUERY_DIM=INNER=512, HEADS=8, DIM_HEAD=64
// ---------------------------------------------------------------------------

typedef __attribute__((ext_vector_type(16))) _Float16 v16h;
typedef __attribute__((ext_vector_type(8)))  _Float16 v8h;
typedef __attribute__((ext_vector_type(8)))  float    v8f;

#define WMMA_F32_F16(a, b, c) \
  __builtin_amdgcn_wmma_f32_16x16x32_f16(false, (a), false, (b), (short)0, (c), false, false)

static constexpr int S_LEN = 4096;
static constexpr int BATCH = 2;
static constexpr int INNER = 512;     // HEADS * DIM_HEAD
static constexpr int QDIM  = 512;
static constexpr int HEADS = 8;
static constexpr int DHEAD = 64;
static constexpr int MTOT  = BATCH * S_LEN;  // 8192

// ---------------------------------------------------------------------------
// Fragment loader: rows-of-K source -> 16-bit 16x32 WMMA operand layout.
// Two contiguous 16-byte loads per lane.
// ---------------------------------------------------------------------------
__device__ __forceinline__ v16h load_frag_nk(const _Float16* p, int ld)
{
  const int lane = threadIdx.x & 31;
  const int r    = lane & 15;
  const int kh   = (lane >> 4) << 3;           // 0 or 8
  const _Float16* base = p + (size_t)r * (size_t)ld + kh;
  v8h lo = *(const v8h*)(base);                // K = kh .. kh+7
  v8h hi = *(const v8h*)(base + 16);           // K = kh+16 .. kh+23
  v16h a;
#pragma unroll
  for (int i = 0; i < 8; ++i) { a[i] = lo[i]; a[i + 8] = hi[i]; }
  return a;
}

// ---------------------------------------------------------------------------
// f32 -> f16 convert (grid-stride)
// ---------------------------------------------------------------------------
__global__ void f32_to_f16_kernel(const float* __restrict__ in,
                                  _Float16* __restrict__ out, int n)
{
  int i = blockIdx.x * blockDim.x + threadIdx.x;
  int stride = gridDim.x * blockDim.x;
  for (; i < n; i += stride) out[i] = (_Float16)in[i];
}

// ---------------------------------------------------------------------------
// C[M,N](f16) = scale * A[M,K] @ Bm^T, Bm stored [N,K] row-major.
// M=8192, N=K=512. Block = 128 threads (4 waves). Wave tile: 32 x 64
// (2 A frags x 4 B frags -> 8 WMMAs per k-step). Cross-iteration software
// pipeline: step k's WMMAs consume step k-1's fragments while step k's loads
// are in flight -> graduated s_wait_loadcnt, no full drains.
// launch_bounds(128,1): allow full VGPR budget so live ranges aren't squashed.
// vstore==1 writes transposed per batch: Ct[b][col][s]  (for V^T).
// ---------------------------------------------------------------------------
__global__ __launch_bounds__(128, 1) void gemm_nt_f16_kernel(
    const _Float16* __restrict__ A, const _Float16* __restrict__ Bm,
    _Float16* __restrict__ C, float scale, int vstore)
{
  const int wave = threadIdx.x >> 5;
  const int lane = threadIdx.x & 31;
  const int m0 = blockIdx.x * 128 + wave * 32;
  const int n0 = blockIdx.y * 64;

  const _Float16* Ar0 = A + (size_t)m0 * 512;
  const _Float16* Ar1 = A + (size_t)(m0 + 16) * 512;

  v8f acc0[4] = {}, acc1[4] = {};

  // prologue: fragments for k-step 0
  v16h a0 = load_frag_nk(Ar0, 512);
  v16h a1 = load_frag_nk(Ar1, 512);
  v16h b[4];
#pragma unroll
  for (int n = 0; n < 4; ++n)
    b[n] = load_frag_nk(Bm + (size_t)(n0 + 16 * n) * 512, 512);

#pragma unroll 2
  for (int kk = 0; kk < 512; kk += 32) {
    const int kn = (kk + 32) & 511;   // wraps to 0 on last iter (dummy preload)
    // issue next-step loads first
    v16h a0n = load_frag_nk(Ar0 + kn, 512);
    v16h a1n = load_frag_nk(Ar1 + kn, 512);
    v16h bn[4];
#pragma unroll
    for (int n = 0; n < 4; ++n)
      bn[n] = load_frag_nk(Bm + (size_t)(n0 + 16 * n) * 512 + kn, 512);
    // compute with current-step fragments (loaded last iteration)
#pragma unroll
    for (int n = 0; n < 4; ++n) acc0[n] = WMMA_F32_F16(a0, b[n], acc0[n]);
#pragma unroll
    for (int n = 0; n < 4; ++n) acc1[n] = WMMA_F32_F16(a1, b[n], acc1[n]);
    // rotate
    a0 = a0n; a1 = a1n;
#pragma unroll
    for (int n = 0; n < 4; ++n) b[n] = bn[n];
  }

  const int c  = lane & 15;
  const int rb = (lane >> 4) << 3;
  if (vstore == 0) {
#pragma unroll
    for (int n = 0; n < 4; ++n)
#pragma unroll
      for (int g = 0; g < 8; ++g) {
        C[(size_t)(m0 + rb + g) * 512 + (n0 + 16 * n + c)] =
            (_Float16)(acc0[n][g] * scale);
        C[(size_t)(m0 + 16 + rb + g) * 512 + (n0 + 16 * n + c)] =
            (_Float16)(acc1[n][g] * scale);
      }
  } else {
#pragma unroll
    for (int n = 0; n < 4; ++n)
#pragma unroll
      for (int g = 0; g < 8; ++g) {
        int col = n0 + 16 * n + c;
        int row0 = m0 + rb + g;
        int row1 = row0 + 16;
        C[((size_t)(row0 >> 12) * INNER + col) * S_LEN + (row0 & (S_LEN - 1))] =
            (_Float16)(acc0[n][g] * scale);
        C[((size_t)(row1 >> 12) * INNER + col) * S_LEN + (row1 & (S_LEN - 1))] =
            (_Float16)(acc1[n][g] * scale);
      }
  }
}

// ---------------------------------------------------------------------------
// out[M,512](f32) = A[M,512](f16) @ Wo^T + bias   (same pipelined structure)
// ---------------------------------------------------------------------------
__global__ __launch_bounds__(128, 1) void gemm_nt_bias_f32_kernel(
    const _Float16* __restrict__ A, const _Float16* __restrict__ Bm,
    const float* __restrict__ bias, float* __restrict__ C)
{
  const int wave = threadIdx.x >> 5;
  const int lane = threadIdx.x & 31;
  const int m0 = blockIdx.x * 128 + wave * 32;
  const int n0 = blockIdx.y * 64;

  const _Float16* Ar0 = A + (size_t)m0 * 512;
  const _Float16* Ar1 = A + (size_t)(m0 + 16) * 512;

  v8f acc0[4] = {}, acc1[4] = {};

  v16h a0 = load_frag_nk(Ar0, 512);
  v16h a1 = load_frag_nk(Ar1, 512);
  v16h b[4];
#pragma unroll
  for (int n = 0; n < 4; ++n)
    b[n] = load_frag_nk(Bm + (size_t)(n0 + 16 * n) * 512, 512);

#pragma unroll 2
  for (int kk = 0; kk < 512; kk += 32) {
    const int kn = (kk + 32) & 511;
    v16h a0n = load_frag_nk(Ar0 + kn, 512);
    v16h a1n = load_frag_nk(Ar1 + kn, 512);
    v16h bn[4];
#pragma unroll
    for (int n = 0; n < 4; ++n)
      bn[n] = load_frag_nk(Bm + (size_t)(n0 + 16 * n) * 512 + kn, 512);
#pragma unroll
    for (int n = 0; n < 4; ++n) acc0[n] = WMMA_F32_F16(a0, b[n], acc0[n]);
#pragma unroll
    for (int n = 0; n < 4; ++n) acc1[n] = WMMA_F32_F16(a1, b[n], acc1[n]);
    a0 = a0n; a1 = a1n;
#pragma unroll
    for (int n = 0; n < 4; ++n) b[n] = bn[n];
  }

  const int c  = lane & 15;
  const int rb = (lane >> 4) << 3;
#pragma unroll
  for (int n = 0; n < 4; ++n)
#pragma unroll
    for (int g = 0; g < 8; ++g) {
      int col = n0 + 16 * n + c;
      C[(size_t)(m0 + rb + g) * 512 + col]      = acc0[n][g] + bias[col];
      C[(size_t)(m0 + 16 + rb + g) * 512 + col] = acc1[n][g] + bias[col];
    }
}

// ---------------------------------------------------------------------------
// Flash attention. Q pre-scaled by DIM_HEAD^-0.5.
//   qh, kh : [B*S, INNER] f16 row-major
//   vt     : [B, INNER, S] f16  (V transposed per batch -> per-head N x K)
//   ctx    : [B*S, INNER] f16
// Grid: (B*H, S/64). Block: 128 threads = 4 waves; wave owns 16 Q rows.
// Per 64-wide KV tile:
//   1. load ALL 8 K fragments (simultaneously live -> graduated waits),
//      then 8 QK^T WMMAs
//   2. issue all 8 V-fragment loads (overlap the softmax)
//   3. global_prefetch_b8 of next K tile
//   4. online softmax (v_exp VALU) while V loads fly
//   5. P -> LDS -> A-fragment reshape, PV WMMAs on landed V fragments
// ---------------------------------------------------------------------------
__global__ __launch_bounds__(128, 1) void flash_attn_kernel(
    const _Float16* __restrict__ qh, const _Float16* __restrict__ kh,
    const _Float16* __restrict__ vt, _Float16* __restrict__ ctx)
{
  __shared__ _Float16 lds_p[4 * 16 * 64];   // per-wave P staging (8 KB)

  const int wave = threadIdx.x >> 5;
  const int lane = threadIdx.x & 31;
  const int bh = blockIdx.x;
  const int b  = bh >> 3;
  const int h  = bh & 7;
  const int q0 = blockIdx.y * 64 + wave * 16;
  const size_t qrow = (size_t)b * S_LEN + q0;

  const _Float16* qp = qh + qrow * INNER + h * DHEAD;
  v16h aq0 = load_frag_nk(qp, INNER);
  v16h aq1 = load_frag_nk(qp + 32, INNER);

  v8f   o[4] = {};
  float m[8], l[8];
#pragma unroll
  for (int g = 0; g < 8; ++g) { m[g] = -INFINITY; l[g] = 0.f; }

  _Float16* pbuf = &lds_p[wave * 16 * 64];
  const _Float16* kbase = kh + (size_t)b * S_LEN * INNER + h * DHEAD;
  const _Float16* vbase = vt + ((size_t)b * INNER + h * DHEAD) * S_LEN;

  const int c  = lane & 15;
  const int rb = (lane >> 4) << 3;

  for (int jt = 0; jt < S_LEN / 64; ++jt) {
    const int j0 = jt * 64;

    // ---- 1. load all K fragments, then 8 QK^T WMMAs ----
    v16h bk0[4], bk1[4];
#pragma unroll
    for (int n = 0; n < 4; ++n) {
      const _Float16* kp = kbase + (size_t)(j0 + 16 * n) * INNER;
      bk0[n] = load_frag_nk(kp, INNER);
      bk1[n] = load_frag_nk(kp + 32, INNER);
    }
    v8f s[4] = {};
#pragma unroll
    for (int n = 0; n < 4; ++n) {
      s[n] = WMMA_F32_F16(aq0, bk0[n], s[n]);
      s[n] = WMMA_F32_F16(aq1, bk1[n], s[n]);
    }

    // ---- 2. V-tile fragment loads now; they overlap the softmax ----
    v16h bv0[4], bv1[4];
#pragma unroll
    for (int n = 0; n < 4; ++n) {
      const _Float16* vp = vbase + (size_t)(16 * n) * S_LEN + j0;
      bv0[n] = load_frag_nk(vp, S_LEN);
      bv1[n] = load_frag_nk(vp + 32, S_LEN);
    }

    // ---- 3. prefetch next K tile (gfx1250 global_prefetch_b8) ----
    if (jt + 1 < S_LEN / 64) {
      const _Float16* knext = kbase + (size_t)(j0 + 64) * INNER;
#pragma unroll
      for (int n = 0; n < 4; ++n)
        __builtin_prefetch(knext + (size_t)(16 * n + (lane & 15)) * INNER, 0, 3);
    }

    // ---- 4. online softmax (rows across 16-lane halves, VGPR idx = row) ----
#pragma unroll
    for (int g = 0; g < 8; ++g) {
      float mx = s[0][g];
#pragma unroll
      for (int n = 1; n < 4; ++n) mx = fmaxf(mx, s[n][g]);
#pragma unroll
      for (int d = 1; d < 16; d <<= 1) mx = fmaxf(mx, __shfl_xor(mx, d, 32));
      const float mn   = fmaxf(m[g], mx);
      const float corr = __expf(m[g] - mn);
      m[g] = mn;
      float rs = 0.f;
#pragma unroll
      for (int n = 0; n < 4; ++n) {
        float pv = __expf(s[n][g] - mn);
        s[n][g] = pv;
        rs += pv;
      }
#pragma unroll
      for (int d = 1; d < 16; d <<= 1) rs += __shfl_xor(rs, d, 32);
      l[g] = l[g] * corr + rs;
#pragma unroll
      for (int n = 0; n < 4; ++n) o[n][g] *= corr;
    }

    // ---- 5. stage P (f16) through LDS: C-layout -> A-fragment layout ----
#pragma unroll
    for (int n = 0; n < 4; ++n)
#pragma unroll
      for (int g = 0; g < 8; ++g)
        pbuf[(rb + g) * 64 + 16 * n + c] = (_Float16)s[n][g];

    v16h ap0 = load_frag_nk(pbuf, 64);
    v16h ap1 = load_frag_nk(pbuf + 32, 64);

    // ---- O += P @ V  (V fragments already in registers) ----
#pragma unroll
    for (int n = 0; n < 4; ++n) {
      o[n] = WMMA_F32_F16(ap0, bv0[n], o[n]);
      o[n] = WMMA_F32_F16(ap1, bv1[n], o[n]);
    }
  }

  // ---- epilogue: normalize and store context (f16) ----
  _Float16* cp = ctx + qrow * INNER + h * DHEAD;
#pragma unroll
  for (int g = 0; g < 8; ++g) {
    const float inv = 1.f / l[g];
#pragma unroll
    for (int n = 0; n < 4; ++n)
      cp[(size_t)(rb + g) * INNER + 16 * n + c] = (_Float16)(o[n][g] * inv);
  }
}

// ---------------------------------------------------------------------------
// Host launcher
// ---------------------------------------------------------------------------
extern "C" void kernel_launch(void* const* d_in, const int* in_sizes, int n_in,
                              void* d_out, int out_size, void* d_ws, size_t ws_size,
                              hipStream_t stream) {
  const float* hs = (const float*)d_in[0];
  const float* Wq = (const float*)d_in[1];
  const float* Wk = (const float*)d_in[2];
  const float* Wv = (const float*)d_in[3];
  const float* Wo = (const float*)d_in[4];
  const float* bo = (const float*)d_in[5];
  float* out = (float*)d_out;

  const size_t ACT_BYTES = (size_t)MTOT * INNER * sizeof(_Float16);  // 8 MB
  const size_t W_BYTES   = (size_t)INNER * QDIM * sizeof(_Float16);  // 512 KB

  char* ws = (char*)d_ws;
  _Float16* hs_h  = (_Float16*)ws;                 ws += ACT_BYTES;
  _Float16* wq_h  = (_Float16*)ws;                 ws += W_BYTES;
  _Float16* wk_h  = (_Float16*)ws;                 ws += W_BYTES;
  _Float16* wv_h  = (_Float16*)ws;                 ws += W_BYTES;
  _Float16* wo_h  = (_Float16*)ws;                 ws += W_BYTES;
  _Float16* q_h   = (_Float16*)ws;                 ws += ACT_BYTES;
  _Float16* k_h   = (_Float16*)ws;                 ws += ACT_BYTES;
  _Float16* vt_h  = (_Float16*)ws;                 ws += ACT_BYTES;  // [B, INNER, S]
  _Float16* ctx_h = (_Float16*)ws;                 ws += ACT_BYTES;

  // Stage 0: precision convert
  f32_to_f16_kernel<<<1024, 256, 0, stream>>>(hs, hs_h, MTOT * QDIM);
  f32_to_f16_kernel<<<256, 256, 0, stream>>>(Wq, wq_h, INNER * QDIM);
  f32_to_f16_kernel<<<256, 256, 0, stream>>>(Wk, wk_h, INNER * QDIM);
  f32_to_f16_kernel<<<256, 256, 0, stream>>>(Wv, wv_h, INNER * QDIM);
  f32_to_f16_kernel<<<256, 256, 0, stream>>>(Wo, wo_h, QDIM * INNER);

  const dim3 blk(128);
  const dim3 ggrid(MTOT / 128, INNER / 64);   // 64 x 8

  // Stage 1: QKV projections (softmax scale folded into Q)
  const float scale = 1.0f / sqrtf((float)DHEAD);  // 0.125
  gemm_nt_f16_kernel<<<ggrid, blk, 0, stream>>>(hs_h, wq_h, q_h, scale, 0);
  gemm_nt_f16_kernel<<<ggrid, blk, 0, stream>>>(hs_h, wk_h, k_h, 1.0f, 0);
  gemm_nt_f16_kernel<<<ggrid, blk, 0, stream>>>(hs_h, wv_h, vt_h, 1.0f, 1);

  // Stage 2: flash attention
  flash_attn_kernel<<<dim3(BATCH * HEADS, S_LEN / 64), blk, 0, stream>>>(
      q_h, k_h, vt_h, ctx_h);

  // Stage 3: output projection + bias (f32 out)
  gemm_nt_bias_f32_kernel<<<ggrid, blk, 0, stream>>>(ctx_h, wo_h, bo, out);

  (void)in_sizes; (void)n_in; (void)out_size; (void)ws_size;
}
